// FeatureAlign_21088289423645
// MI455X (gfx1250) — matrix operand order repaired and checked by reading, not compiled, not run on hardware
//
#include <hip/hip_runtime.h>
#include <hip/hip_bf16.h>

typedef __attribute__((ext_vector_type(16))) _Float16 v16h;
typedef __attribute__((ext_vector_type(8)))  float    v8f;

#define CIN   512
#define COUT  512
#define KTAP  3
#define KDIM  (CIN * KTAP)      // 1536
#define KCH   (KDIM / 32)       // 48 chunks of 32
#define SROW  40                // LDS row stride in halves (80B: 16B-aligned, conflict-free)
#define SBUF  (64 * SROW)       // one S buffer (halves)

// Reduction index ordering: ck = k*512 + c  (tap-major). Each 32-wide chunk q
// has a single tap k = q>>4 and channels c = (q&15)*32 .. +31. No divisions.

// ---------------------------------------------------------------------------
// Pre-pack fp32 weights (Cout, Cin, 3) into f16 WMMA A-fragment order for the
// tap-major reduction: Apack[(ot*48 + q)*32 + lane][elem].
// ---------------------------------------------------------------------------
__global__ __launch_bounds__(256) void prepack_kernel(const float* __restrict__ w,
                                                      _Float16* __restrict__ out) {
    int idx  = blockIdx.x * 256 + threadIdx.x;   // total = 32*48*32*16 = 786432
    int elem = idx & 15;
    int lane = (idx >> 4) & 31;
    int q    = (idx >> 9) % KCH;
    int ot   = (idx >> 9) / KCH;
    int h = lane >> 4;
    int o = ot * 16 + (lane & 15);
    int v = elem >> 1, p = elem & 1;
    int kl = elem + 16 * h;                             // K within chunk (contiguous: k = 16h + elem)
    int ck = q * 32 + kl;                               // tap-major index
    int k  = ck >> 9;                                   // tap
    int c  = ck & 511;                                  // channel
    out[idx] = (_Float16)w[(o * CIN + c) * KTAP + k];
}

// ---------------------------------------------------------------------------
// Offset head: off_out[b,k,t] = ob[k] + off_in[b,k,t]
//              + sum_{c,kk} x[b,c,t+kk-1] * ow[k,c,kk]   (SAME, zero pad)
// ---------------------------------------------------------------------------
__global__ __launch_bounds__(256) void off_conv_kernel(const float* __restrict__ x,
                                                       const float* __restrict__ off_in,
                                                       const float* __restrict__ ow,
                                                       const float* __restrict__ ob,
                                                       float* __restrict__ off_out,
                                                       int T) {
    int idx = blockIdx.x * 256 + threadIdx.x;    // B*T threads
    int b = idx / T, t = idx - b * T;
    const float* xb = x + (size_t)b * CIN * T;
    float a0 = 0.f, a1 = 0.f, a2 = 0.f;
    for (int c = 0; c < CIN; ++c) {
        float xm = (t > 0)     ? xb[c * T + t - 1] : 0.f;
        float x0 =               xb[c * T + t];
        float xp = (t < T - 1) ? xb[c * T + t + 1] : 0.f;
        const float* o0 = ow + c * 3;                    // k = 0
        const float* o1 = ow + (CIN + c) * 3;            // k = 1
        const float* o2 = ow + (2 * CIN + c) * 3;        // k = 2
        a0 += xm * o0[0] + x0 * o0[1] + xp * o0[2];
        a1 += xm * o1[0] + x0 * o1[1] + xp * o1[2];
        a2 += xm * o2[0] + x0 * o2[1] + xp * o2[2];
    }
    off_out[((size_t)b * 3 + 0) * T + t] = a0 + ob[0] + off_in[((size_t)b * 3 + 0) * T + t];
    off_out[((size_t)b * 3 + 1) * T + t] = a1 + ob[1] + off_in[((size_t)b * 3 + 1) * T + t];
    off_out[((size_t)b * 3 + 2) * T + t] = a2 + ob[2] + off_in[((size_t)b * 3 + 2) * T + t];
}

// ---------------------------------------------------------------------------
// Deformable-conv GEMM: y[b,o,t] = relu( sum_{c,k} S[c,k,t]*w[o,c,k] + bias[o] ) * mask[b,t]
// One block (8 waves) = full Cout=512 x 64-column tile for one batch b.
// All varying addresses are 32-bit byte offsets off block-uniform SGPR bases
// (SADDR+voffset form); double-buffered LDS tile, build(q+1) overlaps WMMAs(q).
// ---------------------------------------------------------------------------
__global__ __launch_bounds__(256) void deform_gemm_kernel(const float* __restrict__ x,
                                                          const float* __restrict__ off,
                                                          const _Float16* __restrict__ apack,
                                                          const float* __restrict__ bias,
                                                          const float* __restrict__ mask,
                                                          float* __restrict__ y,
                                                          int T) {
    // per-(k,t) sampling record: {byteOff0, byteOff1, w0, w1} -> one ds_load_b128
    __shared__ __align__(16) int      sSamp[192 * 4];
    __shared__ __align__(16) _Float16 sS[2 * SBUF];

    int nTiles = T >> 6;
    int b  = blockIdx.x / nTiles;
    int t0 = (blockIdx.x - b * nTiles) << 6;
    int tid = threadIdx.x;
    const char* xbc = (const char*)(x + (size_t)b * CIN * T);   // uniform base

    // precompute clamped offsets + masked interpolation weights for this tile
    if (tid < 192) {
        int k = tid / 64, tl = tid - k * 64;
        float pos = (float)(t0 + tl) + (float)(k - 1) + off[((size_t)b * 3 + k) * T + t0 + tl];
        float p0f = floorf(pos);
        int   p0  = (int)p0f;
        float fr  = pos - p0f;
        int   p1  = p0 + 1;
        bool  v0  = (p0 >= 0) && (p0 < T);
        bool  v1  = (p1 >= 0) && (p1 < T);
        sSamp[tid * 4 + 0] = (v0 ? p0 : 0) << 2;          // byte offset, clamped (weight=0 if OOB)
        sSamp[tid * 4 + 1] = (v1 ? p1 : 0) << 2;
        sSamp[tid * 4 + 2] = __float_as_int(v0 ? (1.f - fr) : 0.f);
        sSamp[tid * 4 + 3] = __float_as_int(v1 ? fr : 0.f);
    }

    int lane = tid & 31;
    int wv   = tid >> 5;           // wave id 0..7
    int h    = lane >> 4;
    int ln   = lane & 15;
    int tl_b = tid & 63;           // S-build column
    int klq  = tid >> 6;           // S-build K sub-index 0..3

    const unsigned stepC = (unsigned)T * 16u;      // 4 channels in bytes

    // build one 32xK x 64xT sampled tile for chunk qn into LDS buffer `buf`
    auto buildChunk = [&](int qn, int buf) {
        int k     = qn >> 4;                        // tap: uniform over chunk
        int cbase = (qn & 15) * 32 + klq;           // first channel for this thread
        const int4 sp = *(const int4*)&sSamp[(k * 64 + tl_b) * 4];
        float w0 = __int_as_float(sp.z);
        float w1 = __int_as_float(sp.w);
        unsigned co = (unsigned)cbase * (unsigned)T * 4u;
        unsigned o0 = co + (unsigned)sp.x;          // 32-bit offsets off uniform xbc
        unsigned o1 = co + (unsigned)sp.y;
        _Float16* dst = &sS[buf * SBUF + tl_b * SROW + klq];
        #pragma unroll
        for (int pass = 0; pass < 8; ++pass) {
            float v0 = *(const float*)(xbc + (size_t)o0);
            float v1 = *(const float*)(xbc + (size_t)o1);
            dst[pass * 4] = (_Float16)(v0 * w0 + v1 * w1);
            o0 += stepC;
            o1 += stepC;
        }
    };

    __syncthreads();               // sSamp ready
    buildChunk(0, 0);
    __syncthreads();               // S[0] ready

    v8f acc[4][4] = {};            // [mtile][ngroup], f32 accumulate

    // 32-bit A offset off uniform apack; m-tile = compile-time immediate (+393216B)
    const char* apc = (const char*)apack;           // uniform base
    unsigned aoff = ((unsigned)wv * KCH * 512u + (unsigned)lane * 16u) * 2u;

    for (int q = 0; q < KCH; ++q) {
        int cur = q & 1;

        // issue all 4 A fragments for this chunk (global, streaming)
        v16h af[4];
        #pragma unroll
        for (int mt = 0; mt < 4; ++mt)
            af[mt] = *(const v16h*)(apc + (size_t)aoff + (size_t)mt * (8u * KCH * 512u * 2u));

        // B fragments from current LDS buffer (lane=N%16, half selects K 0..15/16..31)
        v16h bf[4];
        #pragma unroll
        for (int g = 0; g < 4; ++g) {
            union { uint4 u[2]; v16h v; } tmp;
            const uint4* p = (const uint4*)&sS[cur * SBUF + (g * 16 + ln) * SROW + 16 * h];
            tmp.u[0] = p[0];
            tmp.u[1] = p[1];
            bf[g] = tmp.v;
        }

        // overlap: build next chunk's sampled tile into the other buffer
        if (q + 1 < KCH) buildChunk(q + 1, cur ^ 1);

        // 16 WMMAs
        #pragma unroll
        for (int mt = 0; mt < 4; ++mt)
            #pragma unroll
            for (int g = 0; g < 4; ++g)
                acc[mt][g] = __builtin_amdgcn_wmma_f32_16x16x32_f16(
                    false, af[mt], false, bf[g], (short)0, acc[mt][g], false, false);

        aoff += 1024u;             // next chunk's A fragments
        __syncthreads();           // next buffer fully built before anyone reads it
    }

    // ---- epilogue: +bias, relu, *mask, store (C/D layout: M=r+8h, N=ln) ----
    const char* ybc = (const char*)(y + (size_t)b * COUT * T);      // uniform base
    const char* mbc = (const char*)(mask + (size_t)b * T);          // uniform base
    #pragma unroll
    for (int mt = 0; mt < 4; ++mt) {
        int ot = mt * 8 + wv;
        #pragma unroll
        for (int g = 0; g < 4; ++g) {
            unsigned t = (unsigned)(t0 + g * 16 + ln);
            float m = *(const float*)(mbc + (size_t)(t * 4u));
            unsigned ybase = ((unsigned)(ot * 16 + 8 * h) * (unsigned)T + t) * 4u;
            #pragma unroll
            for (int r = 0; r < 8; ++r) {
                int o = ot * 16 + r + 8 * h;
                float val = acc[mt][g][r] + bias[o];
                val = fmaxf(val, 0.f) * m;
                *(float*)(ybc + (size_t)(ybase + (unsigned)r * (unsigned)T * 4u)) = val;
            }
        }
    }
}

// ---------------------------------------------------------------------------
__global__ void mask_out_kernel(const float* __restrict__ m, float* __restrict__ o, int n) {
    int i = blockIdx.x * 256 + threadIdx.x;
    if (i < n) o[i] = (m[i] != 0.f) ? 1.f : 0.f;
}

// ---------------------------------------------------------------------------
extern "C" void kernel_launch(void* const* d_in, const int* in_sizes, int n_in,
                              void* d_out, int out_size, void* d_ws, size_t ws_size,
                              hipStream_t stream) {
    (void)in_sizes; (void)n_in; (void)out_size; (void)ws_size;

    // setup_inputs() dict order: (feats_i, mask_i, offs_i) x3, then (w,b,ow,ob) x2
    const float* feats[3] = {(const float*)d_in[0], (const float*)d_in[3], (const float*)d_in[6]};
    const float* masks[3] = {(const float*)d_in[1], (const float*)d_in[4], (const float*)d_in[7]};
    const float* offs [3] = {(const float*)d_in[2], (const float*)d_in[5], (const float*)d_in[8]};
    const float* w [2] = {(const float*)d_in[9],  (const float*)d_in[13]};
    const float* bi[2] = {(const float*)d_in[10], (const float*)d_in[14]};
    const float* ow[2] = {(const float*)d_in[11], (const float*)d_in[15]};
    const float* ob[2] = {(const float*)d_in[12], (const float*)d_in[16]};

    const int B = 8;
    const int Ts[3] = {2048, 1024, 512};

    // workspace layout
    char* ws = (char*)d_ws;
    const size_t APACK_BYTES = (size_t)32 * KCH * 32 * 16 * sizeof(_Float16);  // 1.5 MB
    _Float16* apack[2] = {(_Float16*)ws, (_Float16*)(ws + APACK_BYTES)};
    float* x1   = (float*)(ws + 2 * APACK_BYTES);                              // up to 33.5 MB
    float* offA = (float*)(ws + 2 * APACK_BYTES + (size_t)B * CIN * 2048 * 4);
    float* offB = offA + (size_t)B * 3 * 2048;

    for (int l = 0; l < 2; ++l)
        prepack_kernel<<<(32 * KCH * 32 * 16) / 256, 256, 0, stream>>>(w[l], apack[l]);

    // output layout: out0,out1,out2 then bool-masks as floats
    float* out = (float*)d_out;
    float* outp[3];
    size_t off_elems = 0;
    for (int i = 0; i < 3; ++i) { outp[i] = out + off_elems; off_elems += (size_t)B * COUT * Ts[i]; }
    float* maskp = out + off_elems;

    for (int i = 0; i < 3; ++i) {
        int T = Ts[i];
        int nBT = B * T;
        // layer 1
        off_conv_kernel<<<nBT / 256, 256, 0, stream>>>(feats[i], offs[i], ow[0], ob[0], offA, T);
        deform_gemm_kernel<<<B * (T / 64), 256, 0, stream>>>(feats[i], offA, apack[0], bi[0],
                                                             masks[i], x1, T);
        // layer 2
        off_conv_kernel<<<nBT / 256, 256, 0, stream>>>(x1, offA, ow[1], ob[1], offB, T);
        deform_gemm_kernel<<<B * (T / 64), 256, 0, stream>>>(x1, offB, apack[1], bi[1],
                                                             masks[i], outp[i], T);
        // boolean mask output
        mask_out_kernel<<<(nBT + 255) / 256, 256, 0, stream>>>(masks[i], maskp, nBT);
        maskp += nBT;
    }
}
